// CPTSelfAttention_2001454760619
// MI455X (gfx1250) — matrix operand order repaired and checked by reading, not compiled, not run on hardware
//
#include <hip/hip_runtime.h>
#include <hip/hip_bf16.h>

// ---------------- problem constants ----------------
#define B_   4
#define S_   2048
#define D_   1024
#define H_   16
#define HD_  64
#define M_   (B_ * S_)        // 8192 rows
#define N1_  (3 * D_)         // 3072 qkv cols
#define SCALING_ 2.0f
#define NEG_INF_ (-1e9f)

typedef __bf16 bf16_t;
typedef __bf16 v8bf  __attribute__((ext_vector_type(8)));
typedef __bf16 v16bf __attribute__((ext_vector_type(16)));
typedef float  v8f   __attribute__((ext_vector_type(8)));

// ---------------- small helpers ----------------
__device__ __forceinline__ float fmaxf2(float a, float b) { return a > b ? a : b; }

// log-domain fake quant: q = sign(x) * s * 2^round(log2(clamp(|x|/s, 2^-255, 1)))
__device__ __forceinline__ bf16_t log_quant1(float x, float s) {
    float a = fabsf(x);
    if (a == 0.0f) return (bf16_t)0.0f;
    float e = rintf(log2f(a / s));
    e = fminf(fmaxf(e, -255.0f), 0.0f);
    float q = copysignf(s * exp2f(e), x);
    return (bf16_t)q;
}

__device__ __forceinline__ v8f wmma_bf16(v16bf a, v16bf b, v8f c) {
    return __builtin_amdgcn_wmma_f32_16x16x32_bf16(false, a, false, b,
                                                   (short)0, c, false, false);
}

// ---------------- utility kernels ----------------
__global__ void zero_u32(unsigned int* p, int n) {
    int i = blockIdx.x * blockDim.x + threadIdx.x;
    if (i < n) p[i] = 0u;
}

// per-column |max| of row-major [Mrows x Ncols] fp32; Ncols % 256 == 0, Mrows % 64 == 0
__global__ void absmax_cols(const float* __restrict__ x, unsigned int* __restrict__ sbits,
                            int Ncols) {
    int col = blockIdx.x * 256 + threadIdx.x;
    long r0 = (long)blockIdx.y * 64;
    float m = 0.0f;
    for (int r = 0; r < 64; ++r)
        m = fmaxf2(m, fabsf(x[(r0 + r) * Ncols + col]));
    atomicMax(&sbits[col], __float_as_uint(m));   // non-negative floats: uint order == float order
}

// quantize activations per-column (last-dim channel)
__global__ void quantize_act(const float* __restrict__ x, const unsigned int* __restrict__ sbits,
                             bf16_t* __restrict__ xq, int total, int Ncols) {
    int i = blockIdx.x * 256 + threadIdx.x;
    if (i >= total) return;
    float s = fmaxf(__uint_as_float(sbits[i % Ncols]), 1e-8f);
    xq[i] = log_quant1(x[i], s);
}

// quantize weights per output row: one block per row of W[N x K]
__global__ void __launch_bounds__(256)
quantize_weight(const float* __restrict__ W, bf16_t* __restrict__ Wq, int K) {
    __shared__ float red[256];
    long n = blockIdx.x;
    const float* row = W + n * K;
    float m = 0.0f;
    for (int k = threadIdx.x; k < K; k += 256) m = fmaxf2(m, fabsf(row[k]));
    red[threadIdx.x] = m;
    __syncthreads();
    for (int off = 128; off > 0; off >>= 1) {
        if ((int)threadIdx.x < off)
            red[threadIdx.x] = fmaxf2(red[threadIdx.x], red[threadIdx.x + off]);
        __syncthreads();
    }
    float s = fmaxf(red[0], 1e-8f);
    for (int k = threadIdx.x; k < K; k += 256) Wq[n * K + k] = log_quant1(row[k], s);
}

// LoRA down-projection: t[m, r] = sum_k x[m,k] * lA[k,r]  (r < 16)
__global__ void lora_down(const float* __restrict__ x, const float* __restrict__ lA,
                          float* __restrict__ t, int K) {
    int i = blockIdx.x * 256 + threadIdx.x;        // i = m*16 + r
    int m = i >> 4, r = i & 15;
    const float* xr = x + (long)m * K;
    float acc = 0.0f;
    for (int k = 0; k < K; ++k) acc += xr[k] * lA[k * 16 + r];
    t[i] = acc;
}

// out[m,n] += bias[n] + SCALING * sum_r t[m,r] * lB[r,n]
__global__ void epilogue_bias_lora(float* __restrict__ out, const float* __restrict__ bias,
                                   const float* __restrict__ t, const float* __restrict__ lB,
                                   int Ncols, int total) {
    int i = blockIdx.x * 256 + threadIdx.x;
    if (i >= total) return;
    int m = i / Ncols, n = i % Ncols;
    const float* tm = t + m * 16;
    float l = 0.0f;
#pragma unroll
    for (int r = 0; r < 16; ++r) l += tm[r] * lB[r * Ncols + n];
    out[i] = out[i] + bias[n] + SCALING_ * l;
}

// ---------------- WMMA GEMM: C[M,N] = A[M,K] * Bw[N,K]^T ----------------
// A-fragment layout (16-bit, 16x32): lanes 0-15 hold row l16, K = [k0..k0+15];
// lanes 16-31 hold K = [k0+16..k0+31].
__device__ __forceinline__ v16bf load_a_frag(const bf16_t* __restrict__ A, int lda,
                                             int row16, int k0, int l16, int hf) {
    const bf16_t* p0 = A + (long)(row16 + l16) * lda + k0 + hf * 16;
    v8bf a0 = *(const v8bf*)p0;
    v8bf a1 = *(const v8bf*)(p0 + 8);
    v16bf a;
#pragma unroll
    for (int i = 0; i < 8; ++i) { a[i] = a0[i]; a[i + 8] = a1[i]; }
    return a;
}

// B-fragment (32x16): lane's column n = col16+l16 maps to row n of Bw; lanes 0-15 hold
// K=[k0..k0+15], lanes 16-31 K=[k0+16..k0+31] -> contiguous 32B from Bw row.
__device__ __forceinline__ v16bf load_b_frag(const bf16_t* __restrict__ Bw, int ldb,
                                             int col16, int k0, int l16, int hf) {
    return *(const v16bf*)(Bw + (long)(col16 + l16) * ldb + k0 + hf * 16);
}

__global__ void __launch_bounds__(256)
gemm_bf16_wmma(const bf16_t* __restrict__ A, const bf16_t* __restrict__ Bw,
               float* __restrict__ C, int Nc, int K) {
    int lane = threadIdx.x & 31;
    int l16 = lane & 15, hf = lane >> 4;
    int wave = threadIdx.x >> 5;
    int wm = wave & 1, wn = wave >> 1;            // 2 waves along M, 4 along N
    int mbase = blockIdx.x * 64 + wm * 32;
    int nbase = blockIdx.y * 128 + wn * 32;

    v8f zero8;
#pragma unroll
    for (int e = 0; e < 8; ++e) zero8[e] = 0.0f;

    v8f acc[2][2];
#pragma unroll
    for (int i = 0; i < 2; ++i)
#pragma unroll
        for (int j = 0; j < 2; ++j) acc[i][j] = zero8;

    for (int k0 = 0; k0 < K; k0 += 32) {
        v16bf af[2], bfr[2];
#pragma unroll
        for (int i = 0; i < 2; ++i) af[i] = load_a_frag(A, K, mbase + i * 16, k0, l16, hf);
#pragma unroll
        for (int j = 0; j < 2; ++j) bfr[j] = load_b_frag(Bw, K, nbase + j * 16, k0, l16, hf);
        if (k0 + 32 < K) {
            __builtin_prefetch(A + (long)(mbase + l16) * K + k0 + 32, 0, 1);
            __builtin_prefetch(Bw + (long)(nbase + l16) * K + k0 + 32, 0, 1);
        }
#pragma unroll
        for (int i = 0; i < 2; ++i)
#pragma unroll
            for (int j = 0; j < 2; ++j)
                acc[i][j] = wmma_bf16(af[i], bfr[j], acc[i][j]);
    }

    // C layout: element e of acc <-> row mbase+i*16+e+8*hf, col nbase+j*16+l16
#pragma unroll
    for (int i = 0; i < 2; ++i)
#pragma unroll
        for (int j = 0; j < 2; ++j) {
            int col = nbase + j * 16 + l16;
#pragma unroll
            for (int e = 0; e < 8; ++e) {
                int row = mbase + i * 16 + e + 8 * hf;
                C[(long)row * Nc + col] = acc[i][j][e];
            }
        }
}

// ---------------- flash attention (causal), bf16 WMMA ----------------
// grid = B*H*(S/64); block = 128 (4 waves); wave w owns 16 query rows.
__global__ void __launch_bounds__(128)
flash_attn(const float* __restrict__ qkv, float* __restrict__ outp) {
    __shared__ alignas(64) bf16_t Kt[32][64];        // key tile, row-major [k][hd]
    __shared__ alignas(64) bf16_t Vt[64][32];        // value tile TRANSPOSED [hd][k]
    __shared__ alignas(64) bf16_t Pt[4][16][32];     // per-wave P staging

    int tid = threadIdx.x;
    int lane = tid & 31, l16 = lane & 15, hf = lane >> 4;
    int w = tid >> 5;
    int nqb = S_ / 64;
    int qb = blockIdx.x % nqb;
    int bh = blockIdx.x / nqb;
    int b = bh / H_, h = bh % H_;
    int s0 = qb * 64 + w * 16;

    // load + convert this wave's Q tile (16 x 64) into two K=32 A-fragments
    v16bf qf[2];
#pragma unroll
    for (int c = 0; c < 2; ++c) {
        const float* qrow = qkv + (long)(b * S_ + s0 + l16) * N1_ + h * HD_ + c * 32;
        v16bf a;
#pragma unroll
        for (int i = 0; i < 8; ++i) {
            a[i]     = (bf16_t)qrow[hf * 16 + i];
            a[i + 8] = (bf16_t)qrow[hf * 16 + 8 + i];
        }
        qf[c] = a;
    }

    v8f zero8;
#pragma unroll
    for (int e = 0; e < 8; ++e) zero8[e] = 0.0f;
    v8f O[4];
#pragma unroll
    for (int f = 0; f < 4; ++f) O[f] = zero8;
    float mrow[8], lrow[8];
#pragma unroll
    for (int e = 0; e < 8; ++e) { mrow[e] = -3.0e38f; lrow[e] = 0.0f; }

    int jmax = 2 * qb + 1;                       // causal: process k blocks up to q block end
    for (int j = 0; j <= jmax; ++j) {
        int ks = j * 32;
        // cooperative stage: K tile (bf16) and V tile (bf16, transposed)
        for (int i = tid; i < 32 * 64; i += 128) {
            int r = i >> 6, c = i & 63;
            long base = (long)(b * S_ + ks + r) * N1_ + h * HD_ + c;
            Kt[r][c] = (bf16_t)qkv[base + D_];
            Vt[c][r] = (bf16_t)qkv[base + 2 * D_];
        }
        __syncthreads();

        // scores: two 16x16 sub-tiles along k
        v8f sfrag[2];
#pragma unroll
        for (int jj = 0; jj < 2; ++jj) {
            v8f s = zero8;
#pragma unroll
            for (int c = 0; c < 2; ++c) {
                v16bf kb = *(const v16bf*)(&Kt[jj * 16 + l16][c * 32 + hf * 16]);
                s = wmma_bf16(qf[c], kb, s);
            }
#pragma unroll
            for (int e = 0; e < 8; ++e) {
                int rq = s0 + e + 8 * hf;
                int ck = ks + jj * 16 + l16;
                float v = s[e] * 0.125f;           // 1/sqrt(64)
                if (ck > rq) v += NEG_INF_;        // causal mask
                s[e] = v;
            }
            sfrag[jj] = s;
        }

        // online softmax: per-row max across 16 lanes of each half-wave
        float alpha[8];
#pragma unroll
        for (int e = 0; e < 8; ++e) {
            float v = fmaxf2(sfrag[0][e], sfrag[1][e]);
            for (int d = 1; d < 16; d <<= 1) v = fmaxf2(v, __shfl_xor(v, d, 32));
            float mn = fmaxf2(mrow[e], v);
            alpha[e] = __expf(mrow[e] - mn);
            mrow[e] = mn;
        }
#pragma unroll
        for (int jj = 0; jj < 2; ++jj)
#pragma unroll
            for (int e = 0; e < 8; ++e) {
                float p = __expf(sfrag[jj][e] - mrow[e]);
                sfrag[jj][e] = p;
                Pt[w][e + 8 * hf][jj * 16 + l16] = (bf16_t)p;   // C-layout -> LDS
            }
#pragma unroll
        for (int e = 0; e < 8; ++e) {
            float rs = sfrag[0][e] + sfrag[1][e];
            for (int d = 1; d < 16; d <<= 1) rs += __shfl_xor(rs, d, 32);
            lrow[e] = lrow[e] * alpha[e] + rs;
#pragma unroll
            for (int f = 0; f < 4; ++f) O[f][e] *= alpha[e];
        }

        // reload P in A-fragment layout from LDS (wave-local; LDS ops are in-order)
        v16bf pa;
        {
            const bf16_t* pr = &Pt[w][l16][hf * 16];
#pragma unroll
            for (int i = 0; i < 8; ++i) { pa[i] = pr[i]; pa[i + 8] = pr[8 + i]; }
        }
        // O += P(16x32) @ V(32x64): B-fragment is contiguous row of transposed V tile
#pragma unroll
        for (int f = 0; f < 4; ++f) {
            v16bf vb = *(const v16bf*)(&Vt[f * 16 + l16][hf * 16]);
            O[f] = wmma_bf16(pa, vb, O[f]);
        }
        __syncthreads();
    }

    // normalize and write out: out[b, s, h*64 + hd]
#pragma unroll
    for (int f = 0; f < 4; ++f)
#pragma unroll
        for (int e = 0; e < 8; ++e) {
            int rq = s0 + e + 8 * hf;
            outp[(long)(b * S_ + rq) * D_ + h * HD_ + f * 16 + l16] = O[f][e] / lrow[e];
        }
}

// ---------------- host-side orchestration ----------------
extern "C" void kernel_launch(void* const* d_in, const int* in_sizes, int n_in,
                              void* d_out, int out_size, void* d_ws, size_t ws_size,
                              hipStream_t stream) {
    const float* x       = (const float*)d_in[0];   // hidden_states [B,S,D]
    const float* mask    = (const float*)d_in[1];   // causal mask (applied analytically)
    const float* W_attn  = (const float*)d_in[2];
    const float* b_attn  = (const float*)d_in[3];
    const float* lA_attn = (const float*)d_in[4];
    const float* lB_attn = (const float*)d_in[5];
    const float* W_proj  = (const float*)d_in[6];
    const float* b_proj  = (const float*)d_in[7];
    const float* lA_proj = (const float*)d_in[8];
    const float* lB_proj = (const float*)d_in[9];
    float* out = (float*)d_out;
    (void)mask; (void)in_sizes; (void)n_in; (void)out_size; (void)ws_size;

    char* ws = (char*)d_ws;
    size_t off = 0;
    auto alloc = [&](size_t bytes) -> void* {
        void* p = ws + off;
        off += (bytes + 255) & ~(size_t)255;
        return p;
    };
    unsigned int* sx1 = (unsigned int*)alloc((size_t)D_ * 4);
    unsigned int* sx2 = (unsigned int*)alloc((size_t)D_ * 4);
    float* t1   = (float*) alloc((size_t)M_ * 16 * 4);
    float* t2   = (float*) alloc((size_t)M_ * 16 * 4);
    bf16_t* xq1 = (bf16_t*)alloc((size_t)M_ * D_ * 2);
    bf16_t* xq2 = (bf16_t*)alloc((size_t)M_ * D_ * 2);
    bf16_t* Wq1 = (bf16_t*)alloc((size_t)N1_ * D_ * 2);
    bf16_t* Wq2 = (bf16_t*)alloc((size_t)D_ * D_ * 2);
    float* qkv  = (float*) alloc((size_t)M_ * N1_ * 4);
    float* aout = (float*) alloc((size_t)M_ * D_ * 4);

    // ---- stage 1: QKV = cpt_linear(x, W_attn, b_attn, lA_attn, lB_attn) ----
    zero_u32<<<(D_ + 255) / 256, 256, 0, stream>>>(sx1, D_);
    zero_u32<<<(D_ + 255) / 256, 256, 0, stream>>>(sx2, D_);
    absmax_cols<<<dim3(D_ / 256, M_ / 64), 256, 0, stream>>>(x, sx1, D_);
    quantize_act<<<(M_ * D_) / 256, 256, 0, stream>>>(x, sx1, xq1, M_ * D_, D_);
    quantize_weight<<<N1_, 256, 0, stream>>>(W_attn, Wq1, D_);
    lora_down<<<(M_ * 16) / 256, 256, 0, stream>>>(x, lA_attn, t1, D_);
    gemm_bf16_wmma<<<dim3(M_ / 64, N1_ / 128), 256, 0, stream>>>(xq1, Wq1, qkv, N1_, D_);
    epilogue_bias_lora<<<(M_ * N1_) / 256, 256, 0, stream>>>(qkv, b_attn, t1, lB_attn,
                                                             N1_, M_ * N1_);

    // ---- stage 2: causal multi-head attention ----
    flash_attn<<<B_ * H_ * (S_ / 64), 128, 0, stream>>>(qkv, aout);

    // ---- stage 3: out = cpt_linear(attn_out, W_proj, b_proj, lA_proj, lB_proj) ----
    absmax_cols<<<dim3(D_ / 256, M_ / 64), 256, 0, stream>>>(aout, sx2, D_);
    quantize_act<<<(M_ * D_) / 256, 256, 0, stream>>>(aout, sx2, xq2, M_ * D_, D_);
    quantize_weight<<<D_, 256, 0, stream>>>(W_proj, Wq2, D_);
    lora_down<<<(M_ * 16) / 256, 256, 0, stream>>>(aout, lA_proj, t2, D_);
    gemm_bf16_wmma<<<dim3(M_ / 64, D_ / 128), 256, 0, stream>>>(xq2, Wq2, out, D_, D_);
    epilogue_bias_lora<<<(M_ * D_) / 256, 256, 0, stream>>>(out, b_proj, t2, lB_proj,
                                                            D_, M_ * D_);
}
